// MixOfExperts_16844861734991
// MI455X (gfx1250) — matrix operand order repaired and checked by reading, not compiled, not run on hardware
//
#include <hip/hip_runtime.h>
#include <math.h>

// ---------------------------------------------------------------------------
// CDNA5 (gfx1250) WMMA types
// ---------------------------------------------------------------------------
typedef __attribute__((ext_vector_type(16))) __bf16 v16bf;
typedef __attribute__((ext_vector_type(8)))  float  v8f;

union FragBF {
    v16bf v;
    uint4 q[2];
};

// ---------------------------------------------------------------------------
// Model constants
// ---------------------------------------------------------------------------
#define BB     64
#define TT     512
#define CC     7
#define SS     (BB * CC)     // 448 sequences
#define DMODEL 256
#define HEADS  8
#define DHEAD  32
#define DFFN   1024
#define PREDN  96
#define NEXP   3
#define LMAX   65            // max sequence length (Pn for p=8/4)

// ---------------------------------------------------------------------------
// bf16 <-> f32 helpers (RNE)
// ---------------------------------------------------------------------------
__device__ __forceinline__ unsigned short f2bf(float f) {
    unsigned int u = __float_as_uint(f);
    unsigned int r = u + 0x7FFFu + ((u >> 16) & 1u);
    return (unsigned short)(r >> 16);
}
__device__ __forceinline__ float bf2f(unsigned short s) {
    return __uint_as_float(((unsigned int)s) << 16);
}

__device__ __forceinline__ float gelu_exact(float v) {
    return 0.5f * v * (1.0f + erff(v * 0.70710678118654752f));
}

// sinusoidal positional encoding value PE[n][d], D = 256
__device__ __forceinline__ float pe_val(int n, int d) {
    int i = d & ~1;
    // ln(10000)/256 = 0.0359778920...
    float div = expf(-(float)i * 0.03597789203f);
    float a = (float)n * div;
    return (d & 1) ? cosf(a) : sinf(a);
}

// ---------------------------------------------------------------------------
// fp32 -> bf16 weight conversion
// ---------------------------------------------------------------------------
__global__ void cvt_kernel(const float* __restrict__ s, unsigned short* __restrict__ d, int n) {
    int i = blockIdx.x * 256 + threadIdx.x;
    if (i < n) d[i] = f2bf(s[i]);
}

__global__ void zero_kernel(float* o, int n) {
    int i = blockIdx.x * 256 + threadIdx.x;
    if (i < n) o[i] = 0.0f;
}

// ---------------------------------------------------------------------------
// Gating: |rfft(x,axis=T)| summed over (C,FR), logits = feat @ w_gate,
// top-2 softmax. One block per batch element.
// ---------------------------------------------------------------------------
__global__ void gate_kernel(const float* __restrict__ x,
                            const float* __restrict__ wg,
                            float* __restrict__ gates) {
    __shared__ float xs[TT * CC];     // 3584 floats
    __shared__ float ctab[512];
    __shared__ float stab[512];
    int b = blockIdx.x;
    int t = threadIdx.x;              // 256 threads

    for (int idx = t; idx < TT * CC; idx += 256)
        xs[idx] = x[(size_t)b * (TT * CC) + idx];
    for (int j = t; j < 512; j += 256) {
        float a = 6.283185307179586f * (float)j * (1.0f / 512.0f);
        ctab[j] = cosf(a);
        stab[j] = sinf(a);
    }
    __syncthreads();

    float l0 = 0.f, l1 = 0.f, l2 = 0.f;
    for (int k = t; k < 257; k += 256) {
        float feat = 0.f;
        for (int c = 0; c < CC; ++c) {
            float re = 0.f, im = 0.f;
            for (int tt = 0; tt < TT; ++tt) {
                int idx = (k * tt) & 511;
                float xv = xs[tt * CC + c];
                re += xv * ctab[idx];
                im += xv * stab[idx];
            }
            feat += sqrtf(re * re + im * im);
        }
        l0 += feat * wg[k * 3 + 0];
        l1 += feat * wg[k * 3 + 1];
        l2 += feat * wg[k * 3 + 2];
    }
    __syncthreads();
    ctab[t] = l0; stab[t] = l1; xs[t] = l2;
    __syncthreads();
    for (int off = 128; off; off >>= 1) {
        if (t < off) {
            ctab[t] += ctab[t + off];
            stab[t] += stab[t + off];
            xs[t]   += xs[t + off];
        }
        __syncthreads();
    }
    if (t == 0) {
        float lg[3] = {ctab[0], stab[0], xs[0]};
        int i1 = 0;
        for (int e = 1; e < 3; ++e) if (lg[e] > lg[i1]) i1 = e;
        int i2 = -1;
        for (int e = 0; e < 3; ++e) {
            if (e == i1) continue;
            if (i2 < 0 || lg[e] > lg[i2]) i2 = e;
        }
        float ev = expf(lg[i2] - lg[i1]);       // <= 1
        float g1 = 1.0f / (1.0f + ev);
        float g2 = ev / (1.0f + ev);
        gates[b * 3 + 0] = 0.f;
        gates[b * 3 + 1] = 0.f;
        gates[b * 3 + 2] = 0.f;
        gates[b * 3 + i1] = g1;
        gates[b * 3 + i2] = g2;
    }
}

// ---------------------------------------------------------------------------
// Token builders. tok1[(s,n),d] = sum_j patch[s,n,j]*W1[d,j] + PE[n,d]
//                 tok2[(s,j),d] = sum_n patch[s,n,j]*W2[d,n] + PE[j,d]
// patch[s=(b,c),n,j] = x[b, min(n*8+j, 511), c]
// ---------------------------------------------------------------------------
__global__ void tok1_kernel(const float* __restrict__ x, const float* __restrict__ W1,
                            unsigned short* __restrict__ tok, int Pn, int p) {
    int row = blockIdx.x;             // s*Pn + n
    int d = threadIdx.x;              // 0..255
    int n = row % Pn;
    int s = row / Pn;
    int b = s / CC, c = s - b * CC;
    float acc = 0.f;
    for (int j = 0; j < p; ++j) {
        int tt = n * 8 + j; if (tt > TT - 1) tt = TT - 1;
        acc += x[((size_t)b * TT + tt) * CC + c] * W1[d * p + j];
    }
    acc += pe_val(n, d);
    tok[(size_t)row * DMODEL + d] = f2bf(acc);
}

__global__ void tok2_kernel(const float* __restrict__ x, const float* __restrict__ W2,
                            unsigned short* __restrict__ tok, int Pn, int p) {
    int row = blockIdx.x;             // s*p + j
    int d = threadIdx.x;
    int j = row % p;
    int s = row / p;
    int b = s / CC, c = s - b * CC;
    float acc = 0.f;
    for (int n = 0; n < Pn; ++n) {
        int tt = n * 8 + j; if (tt > TT - 1) tt = TT - 1;
        acc += x[((size_t)b * TT + tt) * CC + c] * W2[d * Pn + n];
    }
    acc += pe_val(j, d);
    tok[(size_t)row * DMODEL + d] = f2bf(acc);
}

// ---------------------------------------------------------------------------
// WMMA bf16 GEMM:  C[M,N] = A[M,K] * W[N,K]^T + bias[N]   (epi=1 -> GELU)
// Macro-tile 64(M) x 64(N), K-step 64, LDS double-buffered (one barrier per
// K-step).  Block = 256 threads = 8 waves arranged 4(M) x 2(N); each wave
// owns TWO 16x16 output tiles (N and N+16) -> A-fragment reuse, and performs
// 4 x V_WMMA_F32_16X16X32_BF16 per barrier.
// Per-lane fragment layouts used by this kernel (A and B share one K order):
//   A (16-bit 16x32): lanes 0-15 -> M=lane, K 0..15;
//                     lanes 16-31 -> M=lane-16, K 16..31
//   B (16-bit 32x16): lanes 0-15 -> N=lane, K 0..15; lanes 16-31 -> K 16..31
//   C/D f32 16x16:    lanes 0-15 -> N=lane, M=r;  lanes 16-31 -> M=8+r
// ---------------------------------------------------------------------------
#define GM 64
#define GN 64
#define GK 64

__global__ __launch_bounds__(256)
void gemm_bf16_kernel(const unsigned short* __restrict__ A,
                      const unsigned short* __restrict__ W,
                      const float* __restrict__ bias,
                      unsigned short* __restrict__ C,
                      int M, int N, int K, int epi) {
    __shared__ __align__(16) unsigned short As[2][GM][GK];
    __shared__ __align__(16) unsigned short Bs[2][GN][GK];

    const int tile_n0 = blockIdx.x * GN;
    const int tile_m0 = blockIdx.y * GM;
    const int tid  = threadIdx.x;
    const int wave = tid >> 5;
    const int lane = tid & 31;
    const int wm = (wave & 3) << 4;       // 0,16,32,48
    const int wn = (wave >> 2) << 5;      // 0 or 32 (wave owns wn and wn+16)

    v8f acc0 = {0.f, 0.f, 0.f, 0.f, 0.f, 0.f, 0.f, 0.f};
    v8f acc1 = {0.f, 0.f, 0.f, 0.f, 0.f, 0.f, 0.f, 0.f};

    const int lr = tid >> 2;              // load row 0..63
    const int lc = (tid & 3) * 16;        // load col (16 elems = 2x uint4)

    const unsigned short* aptr = A + (size_t)(tile_m0 + lr) * K + lc;
    const unsigned short* wptr = W + (size_t)(tile_n0 + lr) * K + lc;

    // stage one K-step tile pair into LDS buffer `buf`
    auto load_tile = [&](int buf, int k0) {
        *(uint4*)&As[buf][lr][lc]     = *(const uint4*)(aptr + k0);
        *(uint4*)&As[buf][lr][lc + 8] = *(const uint4*)(aptr + k0 + 8);
        *(uint4*)&Bs[buf][lr][lc]     = *(const uint4*)(wptr + k0);
        *(uint4*)&Bs[buf][lr][lc + 8] = *(const uint4*)(wptr + k0 + 8);
        if (k0 + GK < K) {
            __builtin_prefetch(aptr + k0 + GK, 0, 3);
            __builtin_prefetch(wptr + k0 + GK, 0, 3);
        }
    };

    load_tile(0, 0);
    int cur = 0;
    for (int k0 = 0; k0 < K; k0 += GK) {
        __syncthreads();                          // tile `cur` ready
        if (k0 + GK < K) load_tile(cur ^ 1, k0 + GK);  // overlap next loads

        const int am = wm + (lane & 15);
        const int bn = wn + (lane & 15);
#pragma unroll
        for (int ks = 0; ks < 2; ++ks) {          // two K=32 sub-steps
            FragBF fa, fb0, fb1;
            const int ak = ks * 32 + ((lane < 16) ? 0 : 16);
            fa.q[0] = *(const uint4*)&As[cur][am][ak];
            fa.q[1] = *(const uint4*)&As[cur][am][ak + 8];
            const int bk = ks * 32 + ((lane < 16) ? 0 : 16);
            fb0.q[0] = *(const uint4*)&Bs[cur][bn][bk];
            fb0.q[1] = *(const uint4*)&Bs[cur][bn][bk + 8];
            fb1.q[0] = *(const uint4*)&Bs[cur][bn + 16][bk];
            fb1.q[1] = *(const uint4*)&Bs[cur][bn + 16][bk + 8];
            acc0 = __builtin_amdgcn_wmma_f32_16x16x32_bf16(
                false, fa.v, false, fb0.v, (short)0, acc0, false, false);
            acc1 = __builtin_amdgcn_wmma_f32_16x16x32_bf16(
                false, fa.v, false, fb1.v, (short)0, acc1, false, false);
        }
        cur ^= 1;
    }

    const int cn0 = tile_n0 + wn + (lane & 15);
    const int cn1 = cn0 + 16;
    const int rm  = tile_m0 + wm + ((lane < 16) ? 0 : 8);
    const float bv0 = bias[cn0];
    const float bv1 = bias[cn1];
#pragma unroll
    for (int r = 0; r < 8; ++r) {
        float v0 = acc0[r] + bv0;
        float v1 = acc1[r] + bv1;
        if (epi == 1) { v0 = gelu_exact(v0); v1 = gelu_exact(v1); }
        C[(size_t)(rm + r) * N + cn0] = f2bf(v0);
        C[(size_t)(rm + r) * N + cn1] = f2bf(v1);
    }
}

// ---------------------------------------------------------------------------
// Fused softmax attention, one block per (sequence, head). L <= 65, dh=32.
// ---------------------------------------------------------------------------
__global__ void attn_kernel(const unsigned short* __restrict__ Q,
                            const unsigned short* __restrict__ Kx,
                            const unsigned short* __restrict__ V,
                            unsigned short* __restrict__ O, int L) {
    __shared__ float qs[LMAX * DHEAD];
    __shared__ float ks[LMAX * DHEAD];
    __shared__ float vs[LMAX * DHEAD];
    const int s = blockIdx.x, h = blockIdx.y, t = threadIdx.x;
    const int n = L * DHEAD;
    for (int idx = t; idx < n; idx += 128) {
        int l = idx >> 5, d = idx & 31;
        size_t g = ((size_t)s * L + l) * DMODEL + h * DHEAD + d;
        qs[idx] = bf2f(Q[g]);
        ks[idx] = bf2f(Kx[g]);
        vs[idx] = bf2f(V[g]);
    }
    __syncthreads();
    const int i = t;
    if (i < L) {
        float sc[LMAX];
        float mx = -1e30f;
        for (int j = 0; j < L; ++j) {
            float d = 0.f;
            for (int e = 0; e < DHEAD; ++e) d += qs[i * DHEAD + e] * ks[j * DHEAD + e];
            d *= 0.17677669529663687f;      // 1/sqrt(32)
            sc[j] = d;
            mx = fmaxf(mx, d);
        }
        float o[DHEAD];
        for (int e = 0; e < DHEAD; ++e) o[e] = 0.f;
        float den = 0.f;
        for (int j = 0; j < L; ++j) {
            float w = expf(sc[j] - mx);
            den += w;
            for (int e = 0; e < DHEAD; ++e) o[e] += w * vs[j * DHEAD + e];
        }
        float inv = 1.0f / den;
        for (int e = 0; e < DHEAD; ++e)
            O[((size_t)s * L + i) * DMODEL + h * DHEAD + e] = f2bf(o[e] * inv);
    }
}

// ---------------------------------------------------------------------------
// LayerNorm over D=256 with optional residual add: out = LN(a (+ b)) * g + beta
// One block (256 threads) per row. In-place safe.
// ---------------------------------------------------------------------------
__global__ void ln_kernel(const unsigned short* __restrict__ a,
                          const unsigned short* __restrict__ b,
                          const float* __restrict__ g,
                          const float* __restrict__ beta,
                          unsigned short* __restrict__ o) {
    __shared__ float red[DMODEL];
    const int row = blockIdx.x, t = threadIdx.x;
    const size_t idx = (size_t)row * DMODEL + t;
    float x = bf2f(a[idx]);
    if (b) x += bf2f(b[idx]);
    red[t] = x;
    __syncthreads();
    for (int off = 128; off; off >>= 1) {
        if (t < off) red[t] += red[t + off];
        __syncthreads();
    }
    const float m = red[0] * (1.0f / DMODEL);
    __syncthreads();
    const float d = x - m;
    red[t] = d * d;
    __syncthreads();
    for (int off = 128; off; off >>= 1) {
        if (t < off) red[t] += red[t + off];
        __syncthreads();
    }
    const float v = red[0] * (1.0f / DMODEL);
    o[idx] = f2bf(d * rsqrtf(v + 1e-5f) * g[t] + beta[t]);
}

// ---------------------------------------------------------------------------
// Head: dec[b,c,:] = [e1(D,Pn) | e2(D,p)] flattened (d-major), out += gate *
// (dec @ Wh^T + bh) transposed to (B,96,C). Block per (b,c); d split in halves
// so bf16 LDS stays ~21KB.
// ---------------------------------------------------------------------------
__global__ void head_kernel(const unsigned short* __restrict__ e1,
                            const unsigned short* __restrict__ e2,
                            const float* __restrict__ Wh,
                            const float* __restrict__ bh,
                            const float* __restrict__ gates,
                            float* __restrict__ out,
                            int Pn, int p, int eidx) {
    __shared__ unsigned short s1[LMAX * 128];
    __shared__ unsigned short s2[16 * 128];
    const int bc = blockIdx.x;
    const int b = bc / CC, c = bc - b * CC;
    const int t = threadIdx.x;          // 128 threads, 96 active for compute
    const int Ktot = Pn + p;
    float acc = 0.f;
    for (int half = 0; half < 2; ++half) {
        const int d0 = half * 128;
        __syncthreads();
        for (int idx = t; idx < Pn * 128; idx += 128) {
            int n = idx >> 7, dd = idx & 127;
            s1[idx] = e1[((size_t)bc * Pn + n) * DMODEL + d0 + dd];
        }
        for (int idx = t; idx < p * 128; idx += 128) {
            int j = idx >> 7, dd = idx & 127;
            s2[idx] = e2[((size_t)bc * p + j) * DMODEL + d0 + dd];
        }
        __syncthreads();
        if (t < PREDN) {
            const float* wrow = Wh + (size_t)t * DMODEL * Ktot;
            for (int dd = 0; dd < 128; ++dd) {
                const float* w2 = wrow + (size_t)(d0 + dd) * Ktot;
                float a = 0.f;
                for (int n = 0; n < Pn; ++n) a += bf2f(s1[n * 128 + dd]) * w2[n];
                for (int j = 0; j < p; ++j)  a += bf2f(s2[j * 128 + dd]) * w2[Pn + j];
                acc += a;
            }
        }
    }
    if (t < PREDN) {
        float gt = gates[b * 3 + eidx];
        out[((size_t)b * PREDN + t) * CC + c] += gt * (acc + bh[t]);
    }
}

// ---------------------------------------------------------------------------
// Host side
// ---------------------------------------------------------------------------
struct LayerP {
    const float *Wq, *bq, *Wk, *bk, *Wv, *bv, *Wo, *bo;
    const float *Wc1, *bc1, *Wc2, *bc2, *g1, *b1, *g2, *b2;
};
struct EncP { const float *gN, *bN; LayerP L[2]; };
struct ExpP { const float *W1, *W2, *Wh, *bh; EncP enc[2]; };

extern "C" void kernel_launch(void* const* d_in, const int* in_sizes, int n_in,
                              void* d_out, int out_size, void* d_ws, size_t ws_size,
                              hipStream_t stream) {
    (void)n_in; (void)out_size; (void)ws_size;
    auto P = [&](int i) -> const float* { return (const float*)d_in[i]; };

    // ---- map flattened inputs (handle both dict-insertion-order and
    //      jax sorted-key tree flatten orders) ----
    const float *xin = nullptr, *wgate = nullptr;
    ExpP ex[NEXP];
    const bool insertion = (in_sizes[0] == BB * TT * CC);
    if (insertion) {
        xin = P(0); wgate = P(1);
        for (int e = 0; e < NEXP; ++e) {
            int base = 2 + e * 72;
            ex[e].W1 = P(base + 0); ex[e].W2 = P(base + 1);
            for (int en = 0; en < 2; ++en) {
                int eb = base + 2 + en * 34;
                for (int l = 0; l < 2; ++l) {
                    int lb = eb + l * 16;
                    LayerP& L = ex[e].enc[en].L[l];
                    L.Wq = P(lb + 0);  L.bq = P(lb + 1);
                    L.Wk = P(lb + 2);  L.bk = P(lb + 3);
                    L.Wv = P(lb + 4);  L.bv = P(lb + 5);
                    L.Wo = P(lb + 6);  L.bo = P(lb + 7);
                    L.Wc1 = P(lb + 8); L.bc1 = P(lb + 9);
                    L.Wc2 = P(lb + 10); L.bc2 = P(lb + 11);
                    L.g1 = P(lb + 12); L.b1 = P(lb + 13);
                    L.g2 = P(lb + 14); L.b2 = P(lb + 15);
                }
                ex[e].enc[en].gN = P(eb + 32);
                ex[e].enc[en].bN = P(eb + 33);
            }
            ex[e].Wh = P(base + 70); ex[e].bh = P(base + 71);
        }
    } else {
        for (int e = 0; e < NEXP; ++e) {
            int base = e * 72;
            ex[e].W1 = P(base + 0); ex[e].W2 = P(base + 1);
            ex[e].Wh = P(base + 2); ex[e].bh = P(base + 3);
            for (int en = 0; en < 2; ++en) {
                int eb = base + 4 + en * 34;
                ex[e].enc[en].bN = P(eb + 0);
                ex[e].enc[en].gN = P(eb + 1);
                for (int l = 0; l < 2; ++l) {
                    int lb = eb + 2 + l * 16;
                    LayerP& L = ex[e].enc[en].L[l];
                    L.Wc1 = P(lb + 0); L.Wc2 = P(lb + 1);
                    L.Wk = P(lb + 2);  L.Wo = P(lb + 3);
                    L.Wq = P(lb + 4);  L.Wv = P(lb + 5);
                    L.b1 = P(lb + 6);  L.b2 = P(lb + 7);
                    L.bc1 = P(lb + 8); L.bc2 = P(lb + 9);
                    L.bk = P(lb + 10); L.bo = P(lb + 11);
                    L.bq = P(lb + 12); L.bv = P(lb + 13);
                    L.g1 = P(lb + 14); L.g2 = P(lb + 15);
                }
            }
        }
        wgate = P(216); xin = P(217);
    }

    // ---- workspace layout ----
    const size_t ROWMAX = (size_t)SS * LMAX;       // 29120 rows max
    const size_t SZ256  = ROWMAX * DMODEL;          // elements of a D=256 act
    char* ws = (char*)d_ws;
    size_t off = 0;
    unsigned short* WBF = (unsigned short*)(ws + off);
    const size_t LAYER_W = 786432;                  // bf16 elems per layer's 6 matrices
    off += (size_t)NEXP * 2 * 2 * LAYER_W * 2;      // 18.9 MB
    float* GATES = (float*)(ws + off); off += 1024;
    unsigned short* BUF1 = (unsigned short*)(ws + off); off += SZ256 * 2;           // tok1 / e1
    unsigned short* BUF2 = (unsigned short*)(ws + off); off += (size_t)SS*16*DMODEL*2; // tok2 / e2
    unsigned short* TY   = (unsigned short*)(ws + off); off += SZ256 * 2;
    unsigned short* TH2  = (unsigned short*)(ws + off); off += SZ256 * 2;
    unsigned short* R    = (unsigned short*)(ws + off); off += ROWMAX * DFFN * 2;   // QKVO / H
    unsigned short* tQ = R;
    unsigned short* tK = R + SZ256;
    unsigned short* tV = R + 2 * SZ256;
    unsigned short* tO = R + 3 * SZ256;
    unsigned short* tH = R;

    auto cvt = [&](const float* s, unsigned short* d, int n) {
        cvt_kernel<<<(n + 255) / 256, 256, 0, stream>>>(s, d, n);
    };
    auto gemm = [&](const unsigned short* A, const unsigned short* Wm,
                    const float* bias, unsigned short* Cm,
                    int M, int N, int K, int epi) {
        dim3 g(N / GN, M / GM);
        gemm_bf16_kernel<<<g, 256, 0, stream>>>(A, Wm, bias, Cm, M, N, K, epi);
    };

    // ---- zero output, gating ----
    zero_kernel<<<(BB * PREDN * CC + 255) / 256, 256, 0, stream>>>((float*)d_out, BB * PREDN * CC);
    gate_kernel<<<BB, 256, 0, stream>>>(xin, wgate, GATES);

    // ---- convert encoder matrices to bf16 ----
    for (int e = 0; e < NEXP; ++e)
        for (int en = 0; en < 2; ++en)
            for (int l = 0; l < 2; ++l) {
                unsigned short* wb = WBF + ((size_t)((e * 2 + en) * 2 + l)) * LAYER_W;
                const LayerP& L = ex[e].enc[en].L[l];
                cvt(L.Wq,  wb + 0,      DMODEL * DMODEL);
                cvt(L.Wk,  wb + 65536,  DMODEL * DMODEL);
                cvt(L.Wv,  wb + 131072, DMODEL * DMODEL);
                cvt(L.Wo,  wb + 196608, DMODEL * DMODEL);
                cvt(L.Wc1, wb + 262144, DFFN * DMODEL);
                cvt(L.Wc2, wb + 524288, DMODEL * DFFN);
            }

    auto run_enc = [&](unsigned short* buf, int L, const EncP& E,
                       const unsigned short* wenc) {
        const int M = SS * L;
        for (int l = 0; l < 2; ++l) {
            const unsigned short* w = wenc + (size_t)l * LAYER_W;
            const LayerP& lp = E.L[l];
            gemm(buf, w + 0,      lp.bq, tQ, M, DMODEL, DMODEL, 0);
            gemm(buf, w + 65536,  lp.bk, tK, M, DMODEL, DMODEL, 0);
            gemm(buf, w + 131072, lp.bv, tV, M, DMODEL, DMODEL, 0);
            attn_kernel<<<dim3(SS, HEADS), 128, 0, stream>>>(tQ, tK, tV, tO, L);
            gemm(tO, w + 196608, lp.bo, tQ, M, DMODEL, DMODEL, 0);   // proj -> tQ
            ln_kernel<<<M, DMODEL, 0, stream>>>(buf, tQ, lp.g1, lp.b1, TY);
            gemm(TY, w + 262144, lp.bc1, tH, M, DFFN, DMODEL, 1);    // GELU
            gemm(tH, w + 524288, lp.bc2, TH2, M, DMODEL, DFFN, 0);
            ln_kernel<<<M, DMODEL, 0, stream>>>(TY, TH2, lp.g2, lp.b2, buf);
        }
        ln_kernel<<<M, DMODEL, 0, stream>>>(buf, (const unsigned short*)nullptr,
                                            E.gN, E.bN, buf);
    };

    const int PNv[NEXP] = {64, 65, 65};
    const int Pv[NEXP]  = {16, 8, 4};
    for (int e = 0; e < NEXP; ++e) {
        const int Pn = PNv[e], p = Pv[e];
        tok1_kernel<<<SS * Pn, 256, 0, stream>>>(xin, ex[e].W1, BUF1, Pn, p);
        run_enc(BUF1, Pn, ex[e].enc[0], WBF + (size_t)((e * 2 + 0) * 2) * LAYER_W);
        tok2_kernel<<<SS * p, 256, 0, stream>>>(xin, ex[e].W2, BUF2, Pn, p);
        run_enc(BUF2, p, ex[e].enc[1], WBF + (size_t)((e * 2 + 1) * 2) * LAYER_W);
        head_kernel<<<SS, 128, 0, stream>>>(BUF1, BUF2, ex[e].Wh, ex[e].bh,
                                            GATES, (float*)d_out, Pn, p, e);
    }
}
